// gnn_classifier_32049045962971
// MI455X (gfx1250) — compile-verified, matrix operand order checked by reading, not run on hardware
//
#include <hip/hip_runtime.h>
#include <stdint.h>

// ---------------------------------------------------------------------------
// Types for CDNA5 WMMA (wave32): v_wmma_f32_16x16x32_bf16
// ---------------------------------------------------------------------------
typedef __attribute__((ext_vector_type(16))) __bf16 v16bf;
typedef __attribute__((ext_vector_type(8)))  float  v8f;

union FragU { v16bf v; uint4 q[2]; };

__device__ __forceinline__ unsigned short f32_to_bf16(float f) {
  unsigned int u = __float_as_uint(f);
  unsigned int r = u + 0x7FFFu + ((u >> 16) & 1u);  // round-to-nearest-even
  return (unsigned short)(r >> 16);
}
__device__ __forceinline__ float sigm(float x) { return 1.f / (1.f + __expf(-x)); }

// ---------------------------------------------------------------------------
// Elementwise converts / adds
// ---------------------------------------------------------------------------
__global__ void k_cvt_bf16(unsigned short* __restrict__ dst, const float* __restrict__ src, size_t n) {
  size_t i = (size_t)blockIdx.x * blockDim.x + threadIdx.x;
  if (i < n) dst[i] = f32_to_bf16(src[i]);
}

// Convert rows x cols (row stride sld) into rows x dld bf16, zero-padding cols..dld-1.
__global__ void k_cvt_pad_bf16(unsigned short* __restrict__ dst, const float* __restrict__ src,
                               int rows, int cols, int sld, int dld) {
  size_t i = (size_t)blockIdx.x * blockDim.x + threadIdx.x;
  size_t total = (size_t)rows * dld;
  if (i >= total) return;
  int r = (int)(i / dld), c = (int)(i % dld);
  dst[i] = (c < cols) ? f32_to_bf16(src[(size_t)r * sld + c]) : (unsigned short)0;
}

__global__ void k_add(float* __restrict__ dst, const float* __restrict__ a,
                      const float* __restrict__ b, size_t n) {
  size_t i = (size_t)blockIdx.x * blockDim.x + threadIdx.x;
  if (i < n) dst[i] = a[i] + b[i];
}

// ---------------------------------------------------------------------------
// RMSNorm: one block per token row; outputs bf16 (feeds WMMA GEMM)
// ---------------------------------------------------------------------------
__global__ void k_rmsnorm_bf16(unsigned short* __restrict__ out, const float* __restrict__ x,
                               const float* __restrict__ w, int D) {
  __shared__ float red[256];
  int row = blockIdx.x, tid = threadIdx.x;
  const float* xr = x + (size_t)row * D;
  float ss = 0.f;
  for (int c = tid; c < D; c += 256) { float v = xr[c]; ss += v * v; }
  red[tid] = ss; __syncthreads();
  for (int o = 128; o > 0; o >>= 1) { if (tid < o) red[tid] += red[tid + o]; __syncthreads(); }
  float inv = rsqrtf(red[0] / (float)D + 1e-5f);
  for (int c = tid; c < D; c += 256)
    out[(size_t)row * D + c] = f32_to_bf16(xr[c] * inv * w[c]);
}

// ---------------------------------------------------------------------------
// WMMA bf16 GEMM:  C[M,N] = A[M,K] * W[N,K]^T  (+ res[M,N] if res != null)
// Block tile 64(M) x 128(N), 8 waves in a 2x4 grid, each wave owns a 32x32
// patch (2x2 WMMA tiles). K staged 64 at a time, DOUBLE-BUFFERED in LDS:
// next-slab global_load_b128 issued before compute; ds_store into buf^1 after.
// One barrier per slab is sufficient for the 2-buffer rotation.
// LDS row stride 72 halfwords (144 B): conflict-free, 16B-aligned b128 reads.
// Contract (all call sites satisfy it):
//   K % 64 == 0, M % 64 == 0, and W has ceil(N/128)*128 readable rows
//   (garbage rows >= N only feed never-stored accumulator columns).
// ---------------------------------------------------------------------------
#define GBM 64
#define GBN 128
#define GBK 64
#define LSTR 72

__global__ __launch_bounds__(256) void k_gemm_bf16(
    const unsigned short* __restrict__ A, const unsigned short* __restrict__ W,
    float* __restrict__ C, const float* __restrict__ res,
    int M, int N, int K, int ldc) {
  __shared__ unsigned short lsa[2][GBM * LSTR];   // 2 x  9216 B
  __shared__ unsigned short lsb[2][GBN * LSTR];   // 2 x 18432 B

  const int tid  = threadIdx.x;
  const int gm   = blockIdx.y * GBM;
  const int gn   = blockIdx.x * GBN;
  const int lane = tid & 31;
  const int wv   = tid >> 5;
  const int wm   = wv >> 2;        // 0..1 (M waves)
  const int wn   = wv & 3;         // 0..3 (N waves)
  const int m0   = wm * 32, n0 = wn * 32;
  const int mr   = lane & 15;      // row within 16
  const int hl   = lane >> 4;      // K-half select (16-bit WMMA operand layout)

  // staging assignment: 4 threads per row, 8 elements (16B) each, two K halves
  const int arow = tid >> 2;            // 0..63
  const int acq  = (tid & 3) * 8;       // 0,8,16,24 (elements)

  const unsigned short* Arow  = A + (size_t)(gm + arow) * K + acq;
  const unsigned short* Brow0 = W + (size_t)(gn + arow) * K + acq;
  const unsigned short* Brow1 = W + (size_t)(gn + arow + 64) * K + acq;
  const int sOff  = arow * LSTR + acq;
  const int sOffB = (arow + 64) * LSTR + acq;

  // ---- prologue: stage slab 0 into buffer 0 ----
  {
    uint4 a0  = *(const uint4*)(Arow);
    uint4 a1  = *(const uint4*)(Arow + 32);
    uint4 b00 = *(const uint4*)(Brow0);
    uint4 b01 = *(const uint4*)(Brow0 + 32);
    uint4 b10 = *(const uint4*)(Brow1);
    uint4 b11 = *(const uint4*)(Brow1 + 32);
    *(uint4*)(&lsa[0][sOff])       = a0;
    *(uint4*)(&lsa[0][sOff + 32])  = a1;
    *(uint4*)(&lsb[0][sOff])       = b00;
    *(uint4*)(&lsb[0][sOff + 32])  = b01;
    *(uint4*)(&lsb[0][sOffB])      = b10;
    *(uint4*)(&lsb[0][sOffB + 32]) = b11;
  }

  v8f acc[2][2] = {};
  int buf = 0;

  for (int k0 = 0; k0 < K; k0 += GBK) {
    __syncthreads();
    // Issue next-slab loads immediately (overlaps with WMMA below).
    // Last iteration: clamp to 0 -> redundant cached re-read, store is unused.
    int kn = (k0 + GBK < K) ? (k0 + GBK) : 0;
    uint4 a0  = *(const uint4*)(Arow  + kn);
    uint4 a1  = *(const uint4*)(Arow  + kn + 32);
    uint4 b00 = *(const uint4*)(Brow0 + kn);
    uint4 b01 = *(const uint4*)(Brow0 + kn + 32);
    uint4 b10 = *(const uint4*)(Brow1 + kn);
    uint4 b11 = *(const uint4*)(Brow1 + kn + 32);

    const unsigned short* la = lsa[buf];
    const unsigned short* lb = lsb[buf];
#pragma unroll
    for (int ks = 0; ks < GBK; ks += 32) {
      // ISA 16-bit operand layout: lanes 0-15 hold K=0..7 / 16..23,
      // lanes 16-31 hold K=8..15 / 24..31 -> two aligned b128 LDS reads.
      FragU af[2], bfg[2];
#pragma unroll
      for (int i = 0; i < 2; ++i) {
        const unsigned short* p = &la[(m0 + 16 * i + mr) * LSTR + ks + hl * 8];
        af[i].q[0] = *(const uint4*)p;
        af[i].q[1] = *(const uint4*)(p + 16);
      }
#pragma unroll
      for (int j = 0; j < 2; ++j) {
        const unsigned short* p = &lb[(n0 + 16 * j + mr) * LSTR + ks + hl * 8];
        bfg[j].q[0] = *(const uint4*)p;
        bfg[j].q[1] = *(const uint4*)(p + 16);
      }
#pragma unroll
      for (int i = 0; i < 2; ++i)
#pragma unroll
        for (int j = 0; j < 2; ++j)
          acc[i][j] = __builtin_amdgcn_wmma_f32_16x16x32_bf16(
              false, af[i].v, false, bfg[j].v, (short)0, acc[i][j], false, false);
    }

    // Stage the fetched slab into the other buffer (safe: its last readers
    // finished before the barrier at the top of this iteration).
    unsigned short* da = lsa[buf ^ 1];
    unsigned short* db = lsb[buf ^ 1];
    *(uint4*)(&da[sOff])       = a0;
    *(uint4*)(&da[sOff + 32])  = a1;
    *(uint4*)(&db[sOff])       = b00;
    *(uint4*)(&db[sOff + 32])  = b01;
    *(uint4*)(&db[sOffB])      = b10;
    *(uint4*)(&db[sOffB + 32]) = b11;
    buf ^= 1;
  }

  // ---- store (C/D layout: lanes 0-15 M=r, lanes 16-31 M=8+r; N = lane&15) ----
#pragma unroll
  for (int i = 0; i < 2; ++i)
#pragma unroll
    for (int j = 0; j < 2; ++j) {
      int n = gn + n0 + 16 * j + mr;
      if (n < N) {
        int mb = gm + m0 + 16 * i + hl * 8;
#pragma unroll
        for (int r = 0; r < 8; ++r) {
          int m = mb + r;
          if (m < M) {
            float v = acc[i][j][r];
            if (res) v += res[(size_t)m * ldc + n];
            C[(size_t)m * ldc + n] = v;
          }
        }
      }
    }
}

// ---------------------------------------------------------------------------
// Depthwise causal conv (D_CONV=4) + bias + SiLU. xi = columns [0,DI) of xz.
// Writes f32 u (for scan) and bf16 u (for x_proj GEMM).
// ---------------------------------------------------------------------------
__global__ void k_conv_silu(float* __restrict__ u, unsigned short* __restrict__ u_bf,
                            const float* __restrict__ xz, const float* __restrict__ cw,
                            const float* __restrict__ cb, int nseq, int L, int DI, int ldxz) {
  size_t i = (size_t)blockIdx.x * blockDim.x + threadIdx.x;
  size_t total = (size_t)nseq * L * DI;
  if (i >= total) return;
  int c = (int)(i % DI);
  size_t tl = i / DI;
  int t = (int)(tl % L), s = (int)(tl / L);
  float acc = cb[c];
#pragma unroll
  for (int j = 0; j < 4; ++j) {
    int tt = t - 3 + j;
    if (tt >= 0) acc += cw[(size_t)c * 4 + j] * xz[((size_t)s * L + tt) * ldxz + c];
  }
  float v = acc * sigm(acc);
  u[i] = v;
  u_bf[i] = f32_to_bf16(v);
}

// delta = softplus(raw + dt_b)   (in place)
__global__ void k_softplus_bias(float* __restrict__ delta, const float* __restrict__ db,
                                int T, int DI) {
  size_t i = (size_t)blockIdx.x * blockDim.x + threadIdx.x;
  if (i >= (size_t)T * DI) return;
  float v = delta[i] + db[i % DI];
  delta[i] = (v > 20.f) ? v : log1pf(__expf(v));
}

// ---------------------------------------------------------------------------
// Selective scan: thread = (sequence, channel). h[16] in registers.
// Fuses + u*D, * silu(z), and bf16 downconvert (feeds out_proj GEMM).
// ---------------------------------------------------------------------------
__global__ void k_scan(unsigned short* __restrict__ yg_bf, const float* __restrict__ delta,
                       const float* __restrict__ u, const float* __restrict__ xdbl, int ldxd,
                       int dtr, const float* __restrict__ xz, int ldxz, int DI,
                       const float* __restrict__ A_log, const float* __restrict__ Dvec,
                       int nseq, int L) {
  int gid = blockIdx.x * blockDim.x + threadIdx.x;
  int c = gid % DI, s = gid / DI;
  if (s >= nseq) return;
  float Aa[16], h[16];
#pragma unroll
  for (int n = 0; n < 16; ++n) { Aa[n] = -__expf(A_log[(size_t)c * 16 + n]); h[n] = 0.f; }
  float Dv = Dvec[c];
  for (int t = 0; t < L; ++t) {
    size_t row = (size_t)s * L + t;
    float d  = delta[row * DI + c];
    float uu = u[row * DI + c];
    const float* Brow = xdbl + row * ldxd + dtr;
    const float* Crow = Brow + 16;
    float du = d * uu, accy = 0.f;
#pragma unroll
    for (int n = 0; n < 16; ++n) {
      h[n] = __expf(d * Aa[n]) * h[n] + du * Brow[n];
      accy += h[n] * Crow[n];
    }
    float y = accy + uu * Dv;
    float z = xz[row * ldxz + DI + c];
    yg_bf[row * DI + c] = f32_to_bf16(y * (z * sigm(z)));
  }
}

// ---------------------------------------------------------------------------
// Gather segment rows (start from int64 segments tensor) -> f32 + bf16
// ---------------------------------------------------------------------------
__global__ void k_gather_seg(float* __restrict__ dst, unsigned short* __restrict__ dst_bf,
                             const float* __restrict__ x, const long long* __restrict__ seg,
                             int L, int D, int nsegb, int Tseg) {
  int g = blockIdx.y;
  int b = g / nsegb, s2 = g % nsegb;
  int start = (int)seg[((size_t)b * nsegb + s2) * 2 + 0];
  int idx = blockIdx.x * blockDim.x + threadIdx.x;
  if (idx >= Tseg * D) return;
  int t = idx / D, c = idx % D;
  float v = x[((size_t)b * L + start + t) * D + c];
  size_t o = ((size_t)g * Tseg + t) * D + c;
  dst[o] = v;
  dst_bf[o] = f32_to_bf16(v);
}

// ---------------------------------------------------------------------------
// LSTM recurrence: one workgroup per (segment, direction), 320 threads = H.
// gx = precomputed e @ Wih^T (WMMA GEMM). h shared in LDS, c in registers.
// ---------------------------------------------------------------------------
__global__ void k_lstm(float* __restrict__ lo, const float* __restrict__ gx,
                       const float* __restrict__ Whh, const float* __restrict__ bih,
                       const float* __restrict__ bhh, int H, int Tseg, int coloff,
                       int reverse, int ldo) {
  __shared__ float hs[320];
  int g = blockIdx.x, n = threadIdx.x;
  float cc = 0.f;
  hs[n] = 0.f;
  float bi  = bih[n]         + bhh[n];
  float bff = bih[n + H]     + bhh[n + H];
  float bg  = bih[n + 2 * H] + bhh[n + 2 * H];
  float bo  = bih[n + 3 * H] + bhh[n + 3 * H];
  const float* Wi = Whh + (size_t)n * H;
  const float* Wf = Whh + (size_t)(n + H) * H;
  const float* Wg = Whh + (size_t)(n + 2 * H) * H;
  const float* Wo = Whh + (size_t)(n + 3 * H) * H;
  __syncthreads();
  for (int step = 0; step < Tseg; ++step) {
    int t = reverse ? (Tseg - 1 - step) : step;
    const float* gr = gx + ((size_t)g * Tseg + t) * (4 * H);
    float gi = gr[n] + bi, gf = gr[n + H] + bff, gg = gr[n + 2 * H] + bg, go = gr[n + 3 * H] + bo;
    for (int k = 0; k < H; ++k) {
      float hk = hs[k];
      gi += Wi[k] * hk; gf += Wf[k] * hk; gg += Wg[k] * hk; go += Wo[k] * hk;
    }
    cc = sigm(gf) * cc + sigm(gi) * tanhf(gg);
    float hn = sigm(go) * tanhf(cc);
    __syncthreads();
    hs[n] = hn;
    __syncthreads();
    lo[((size_t)g * Tseg + t) * ldo + coloff + n] = hn;
  }
}

// ---------------------------------------------------------------------------
// Attention pooling + layernorm + classifier. One workgroup per segment.
// Writes results[g] and emb[g*C ..] directly into d_out.
// ---------------------------------------------------------------------------
__global__ void k_att_cls(float* __restrict__ results, float* __restrict__ emb,
                          const float* __restrict__ lo, const float* __restrict__ aw,
                          const float* __restrict__ ab, const float* __restrict__ lnw,
                          const float* __restrict__ lnb, const float* __restrict__ cw,
                          const float* __restrict__ cb, int T, int C) {
  __shared__ float attn[64];
  __shared__ float sv[640];
  __shared__ float red[256];
  int g = blockIdx.x, tid = threadIdx.x;
  const float* e = lo + (size_t)g * T * C;

  if (tid < T) {
    float a = ab[0];
    for (int c = 0; c < C; ++c) a += e[(size_t)tid * C + c] * aw[c];
    attn[tid] = fmaxf(a, 0.f);
  }
  __syncthreads();
  if (tid == 0) {  // softmax over tokens (tiny: T=56)
    float mx = -1e30f;
    for (int t = 0; t < T; ++t) mx = fmaxf(mx, attn[t]);
    float s = 0.f;
    for (int t = 0; t < T; ++t) { float v = __expf(attn[t] - mx); attn[t] = v; s += v; }
    float inv = 1.f / s;
    for (int t = 0; t < T; ++t) attn[t] *= inv;
  }
  __syncthreads();
  float scale = rsqrtf((float)T);
  for (int c = tid; c < C; c += 256) {
    float acc = 0.f;
    for (int t = 0; t < T; ++t) acc += e[(size_t)t * C + c] * attn[t];
    sv[c] = acc * scale;
  }
  __syncthreads();
  float lm = 0.f;
  for (int c = tid; c < C; c += 256) lm += sv[c];
  red[tid] = lm; __syncthreads();
  for (int o = 128; o > 0; o >>= 1) { if (tid < o) red[tid] += red[tid + o]; __syncthreads(); }
  float mu = red[0] / (float)C;
  __syncthreads();
  float lv = 0.f;
  for (int c = tid; c < C; c += 256) { float d = sv[c] - mu; lv += d * d; }
  red[tid] = lv; __syncthreads();
  for (int o = 128; o > 0; o >>= 1) { if (tid < o) red[tid] += red[tid + o]; __syncthreads(); }
  float inv = rsqrtf(red[0] / (float)C + 1e-5f);
  __syncthreads();
  float lc = 0.f;
  for (int c = tid; c < C; c += 256) {
    float o2 = (sv[c] - mu) * inv * lnw[c] + lnb[c];
    emb[(size_t)g * C + c] = o2;
    lc += fmaxf(o2, 0.f) * cw[c];
  }
  red[tid] = lc; __syncthreads();
  for (int o = 128; o > 0; o >>= 1) { if (tid < o) red[tid] += red[tid + o]; __syncthreads(); }
  if (tid == 0) results[g] = sigm(red[0] + cb[0]);
}

// ---------------------------------------------------------------------------
// Host-side orchestration
// ---------------------------------------------------------------------------
static inline unsigned cdivu(size_t a, size_t b) { return (unsigned)((a + b - 1) / b); }

struct Ws {
  float *XA, *XB, *XZ, *U, *XDBL, *DELTA, *EG, *GXF, *GXB, *LO, *LOF;
  unsigned short *XN_BF, *U_BF, *DTP_BF, *YG_BF, *WIN_BF, *WX_BF, *WDT_BF, *WOUT_BF, *EG_BF;
};

struct MParams {
  const float *norm_w, *in_proj, *conv_w, *conv_b, *x_proj, *dt_w, *dt_b, *A_log, *Dv, *out_proj;
  int nl, D, DI, dtr, E, KP;   // KP: dt_rank padded to a multiple of 64 (GEMM BK)
};

static inline void gemm_bf16(hipStream_t st, const unsigned short* A, const unsigned short* W,
                             float* C, const float* res, int M, int N, int K, int ldc) {
  dim3 g((unsigned)((N + GBN - 1) / GBN), (unsigned)((M + GBM - 1) / GBM));
  k_gemm_bf16<<<g, 256, 0, st>>>(A, W, C, res, M, N, K, ldc);
}

static const float* run_mamba(hipStream_t st, const Ws& w, const MParams& p,
                              const float* x0, float* bufA, float* bufB, int nseq, int L) {
  const int D = p.D, DI = p.DI, dtr = p.dtr, E = p.E, KP = p.KP;
  const int T = nseq * L;
  const float* cur = x0;
  for (int l = 0; l < p.nl; ++l) {
    float* nxt = (cur == bufA) ? bufB : bufA;
    const float* nw = p.norm_w  + (size_t)l * D;
    const float* ip = p.in_proj + (size_t)l * 2 * DI * D;
    const float* cw = p.conv_w  + (size_t)l * DI * 4;
    const float* cb = p.conv_b  + (size_t)l * DI;
    const float* xp = p.x_proj  + (size_t)l * E * DI;
    const float* dw = p.dt_w    + (size_t)l * DI * dtr;
    const float* db = p.dt_b    + (size_t)l * DI;
    const float* al = p.A_log   + (size_t)l * DI * 16;
    const float* dv = p.Dv      + (size_t)l * DI;
    const float* op = p.out_proj+ (size_t)l * D * DI;

    k_rmsnorm_bf16<<<T, 256, 0, st>>>(w.XN_BF, cur, nw, D);

    size_t n1 = (size_t)2 * DI * D;
    k_cvt_bf16<<<cdivu(n1, 256), 256, 0, st>>>(w.WIN_BF, ip, n1);
    size_t n2 = (size_t)E * DI;
    k_cvt_bf16<<<cdivu(n2, 256), 256, 0, st>>>(w.WX_BF, xp, n2);
    k_cvt_pad_bf16<<<cdivu((size_t)DI * KP, 256), 256, 0, st>>>(w.WDT_BF, dw, DI, dtr, dtr, KP);
    size_t n3 = (size_t)D * DI;
    k_cvt_bf16<<<cdivu(n3, 256), 256, 0, st>>>(w.WOUT_BF, op, n3);

    gemm_bf16(st, w.XN_BF, w.WIN_BF, w.XZ, nullptr, T, 2 * DI, D, 2 * DI);
    k_conv_silu<<<cdivu((size_t)T * DI, 256), 256, 0, st>>>(w.U, w.U_BF, w.XZ, cw, cb, nseq, L, DI, 2 * DI);
    gemm_bf16(st, w.U_BF, w.WX_BF, w.XDBL, nullptr, T, E, DI, E);
    k_cvt_pad_bf16<<<cdivu((size_t)T * KP, 256), 256, 0, st>>>(w.DTP_BF, w.XDBL, T, dtr, E, KP);
    gemm_bf16(st, w.DTP_BF, w.WDT_BF, w.DELTA, nullptr, T, DI, KP, DI);
    k_softplus_bias<<<cdivu((size_t)T * DI, 256), 256, 0, st>>>(w.DELTA, db, T, DI);
    k_scan<<<cdivu((size_t)nseq * DI, 256), 256, 0, st>>>(w.YG_BF, w.DELTA, w.U, w.XDBL, E, dtr,
                                                          w.XZ, 2 * DI, DI, al, dv, nseq, L);
    gemm_bf16(st, w.YG_BF, w.WOUT_BF, nxt, cur, T, D, DI, D);
    cur = nxt;
  }
  return cur;
}

extern "C" void kernel_launch(void* const* d_in, const int* in_sizes, int n_in,
                              void* d_out, int out_size, void* d_ws, size_t ws_size,
                              hipStream_t stream) {
  (void)in_sizes; (void)n_in; (void)out_size; (void)ws_size;

  const float*     emb  = (const float*)d_in[0];
  const long long* segs = (const long long*)d_in[1];

  // dt_rank padded to 128/64 so every GEMM K is a multiple of 64 (BK)
  MParams m1 = { (const float*)d_in[2],  (const float*)d_in[3],  (const float*)d_in[4],
                 (const float*)d_in[5],  (const float*)d_in[6],  (const float*)d_in[7],
                 (const float*)d_in[8],  (const float*)d_in[9],  (const float*)d_in[10],
                 (const float*)d_in[11], 3, 1280, 2560, 80, 112, 128 };
  MParams m2 = { (const float*)d_in[12], (const float*)d_in[13], (const float*)d_in[14],
                 (const float*)d_in[15], (const float*)d_in[16], (const float*)d_in[17],
                 (const float*)d_in[18], (const float*)d_in[19], (const float*)d_in[20],
                 (const float*)d_in[21], 12, 640, 1280, 40, 72, 64 };

  const float* Wih_f = (const float*)d_in[22];
  const float* Whh_f = (const float*)d_in[23];
  const float* bih_f = (const float*)d_in[24];
  const float* bhh_f = (const float*)d_in[25];
  const float* Wih_b = (const float*)d_in[26];
  const float* Whh_b = (const float*)d_in[27];
  const float* bih_b = (const float*)d_in[28];
  const float* bhh_b = (const float*)d_in[29];
  const float* attw  = (const float*)d_in[30];
  const float* attb  = (const float*)d_in[31];
  const float* lnw   = (const float*)d_in[32];
  const float* lnb   = (const float*)d_in[33];
  const float* clsw  = (const float*)d_in[34];
  const float* clsb  = (const float*)d_in[35];

  float* out_results = (float*)d_out;      // (2,4,1) -> 8 floats
  float* out_emb     = out_results + 8;    // (8,640)

  // ---- workspace carve (~65 MB) ----
  char* base = (char*)d_ws; size_t off = 0;
  auto alloc = [&](size_t bytes) -> void* {
    void* pp = base + off; off += (bytes + 255) & ~(size_t)255; return pp;
  };
  Ws w;
  w.XA      = (float*)alloc((size_t)512 * 1280 * 4);
  w.XB      = (float*)alloc((size_t)512 * 1280 * 4);
  w.XZ      = (float*)alloc((size_t)512 * 5120 * 4);
  w.U       = (float*)alloc((size_t)512 * 2560 * 4);
  w.XDBL    = (float*)alloc((size_t)512 * 112 * 4);
  w.DELTA   = (float*)alloc((size_t)512 * 2560 * 4);
  w.EG      = (float*)alloc((size_t)448 * 1280 * 4);
  w.GXF     = (float*)alloc((size_t)448 * 1280 * 4);
  w.GXB     = (float*)alloc((size_t)448 * 1280 * 4);
  w.LO      = (float*)alloc((size_t)448 * 640 * 4);
  w.LOF     = (float*)alloc((size_t)448 * 640 * 4);
  w.XN_BF   = (unsigned short*)alloc((size_t)512 * 1280 * 2);
  w.U_BF    = (unsigned short*)alloc((size_t)512 * 2560 * 2);
  w.DTP_BF  = (unsigned short*)alloc((size_t)512 * 128 * 2);
  w.YG_BF   = (unsigned short*)alloc((size_t)512 * 2560 * 2);
  w.WIN_BF  = (unsigned short*)alloc((size_t)5120 * 1280 * 2);
  // x_proj weight staging: 128 rows allocated (N=112/72 padded to the 128-row
  // block tile so GEMM staging needs no bounds checks; extra rows are never
  // stored because the C-store is guarded by n < N).
  w.WX_BF   = (unsigned short*)alloc((size_t)128 * 2560 * 2);
  w.WDT_BF  = (unsigned short*)alloc((size_t)2560 * 128 * 2);
  w.WOUT_BF = (unsigned short*)alloc((size_t)1280 * 2560 * 2);
  w.EG_BF   = (unsigned short*)alloc((size_t)448 * 1280 * 2);

  // ---- stage 1: x = mamba(emb, m1) + emb ; x = mamba(x, m1) ----
  const float* p1 = run_mamba(stream, w, m1, emb, w.XA, w.XB, 2, 256);
  k_add<<<cdivu((size_t)512 * 1280, 256), 256, 0, stream>>>((float*)p1, p1, emb, (size_t)512 * 1280);
  const float* p2 = run_mamba(stream, w, m1, p1, w.XA, w.XB, 2, 256);

  // ---- gather segments (8 segments x 56 tokens x 1280) ----
  dim3 gg(cdivu((size_t)56 * 1280, 256), 8);
  k_gather_seg<<<gg, 256, 0, stream>>>(w.EG, w.EG_BF, p2, segs, 256, 1280, 4, 56);

  // ---- bidirectional LSTM: input projections via WMMA, then recurrence ----
  size_t nwih = (size_t)1280 * 1280;
  k_cvt_bf16<<<cdivu(nwih, 256), 256, 0, stream>>>(w.WIN_BF, Wih_f, nwih);
  gemm_bf16(stream, w.EG_BF, w.WIN_BF, w.GXF, nullptr, 448, 1280, 1280, 1280);
  k_cvt_bf16<<<cdivu(nwih, 256), 256, 0, stream>>>(w.WIN_BF, Wih_b, nwih);
  gemm_bf16(stream, w.EG_BF, w.WIN_BF, w.GXB, nullptr, 448, 1280, 1280, 1280);
  k_lstm<<<8, 320, 0, stream>>>(w.LO, w.GXF, Whh_f, bih_f, bhh_f, 320, 56, 0,   0, 640);
  k_lstm<<<8, 320, 0, stream>>>(w.LO, w.GXB, Whh_b, bih_b, bhh_b, 320, 56, 320, 1, 640);

  // ---- stage 2: lo = mamba(lo, m2) + lo  (8 sequences of length 56) ----
  const float* p3 = run_mamba(stream, w, m2, w.LO, w.XA, w.XB, 8, 56);
  k_add<<<cdivu((size_t)448 * 640, 256), 256, 0, stream>>>(w.LOF, p3, w.LO, (size_t)448 * 640);

  // ---- attention pool + layernorm + classifier, straight into d_out ----
  k_att_cls<<<8, 256, 0, stream>>>(out_results, out_emb, w.LOF, attw, attb, lnw, lnb,
                                   clsw, clsb, 56, 640);
}
